// HybridSiLUGatedAttention_30666066493561
// MI455X (gfx1250) — matrix operand order repaired and checked by reading, not compiled, not run on hardware
//
#include <hip/hip_runtime.h>
#include <hip/hip_bf16.h>
#include <math.h>

// ---------------- problem constants ----------------
#define D_MODEL   1024
#define NUM_HEADS 16
#define HEAD_DIM  64
#define BATCH     2
#define SEQ       2048
#define ROWS      (BATCH*SEQ)          // 4096
#define ATTN_SCALE 0.125f              // 64^-0.5
#define LN_EPS    1e-5f

typedef __attribute__((ext_vector_type(16))) __bf16 v16bf;
typedef __attribute__((ext_vector_type(8)))  float  v8f;

union FragAB { v16bf v; unsigned int u[8]; };

// A-matrix (16x32 bf16) per-lane K base for packed dword j (ISA 7.12.2)
__device__ __forceinline__ int kbaseA(int j, int hi) { return 2*j + 8*((j>=4) + hi); }
// B-matrix (32x16 bf16) per-lane K base for packed dword j
__device__ __forceinline__ int kbaseB(int j, int hi) { return 2*j + 8*((j>=4) + hi); }

// SiLU with v_rcp_f32 instead of IEEE divide (reference math tolerance is fine)
__device__ __forceinline__ float silu_f(float x) {
  return x * __builtin_amdgcn_rcpf(1.0f + __expf(-x));
}

// ---- CDNA5 async memory->LDS copy (ASYNCcnt-tracked, bypasses VGPRs) ----
// LDS byte offset = low 32 bits of the generic/flat shared address
// (ISA 10.2: LDS aperture maps addr[31:0] directly to LDS).
__device__ __forceinline__ void async_load_b128(void* lds_ptr, const void* gptr) {
  unsigned lds_off = (unsigned)(size_t)lds_ptr;
  unsigned long long ga = (unsigned long long)(size_t)gptr;
  asm volatile("global_load_async_to_lds_b128 %0, %1, off"
               :: "v"(lds_off), "v"(ga) : "memory");
}
__device__ __forceinline__ void wait_asynccnt0() {
  asm volatile("s_wait_asynccnt 0" ::: "memory");
}

// ---------------- elementwise f32 -> bf16 ----------------
__global__ void cvt_f32_bf16_kernel(const float* __restrict__ src,
                                    __bf16* __restrict__ dst, int n) {
  int i = blockIdx.x * blockDim.x + threadIdx.x;
  if (i < n) dst[i] = (__bf16)src[i];
}

// ---------------- mask normalizer: sqrt(max(1, sum_k mask)) ----------------
__global__ void mask_norm_kernel(const int* __restrict__ mask,
                                 float* __restrict__ norm) {
  int row = blockIdx.x * blockDim.x + threadIdx.x;   // 0..ROWS-1 (b*SEQ+q)
  if (row >= ROWS) return;
  const int* mp = mask + (size_t)row * SEQ;
  int s = 0;
  for (int k = 0; k < SEQ; ++k) s += mp[k];
  float f = (float)s;
  if (f < 1.0f) f = 1.0f;
  norm[row] = sqrtf(f);
}

// ---------------- tiled WMMA GEMM core: 128x128 block, 8 waves ----------------
// K-stage = 64 (two WMMA sub-steps per LDS stage -> half the barriers).
// ASTR = 72 bf16 elems (144 B): even (dword pair reads) and 16B-multiple
// (async B128 chunks land aligned).
#define ASTR 72

__device__ __forceinline__ void gemm_core_128x128(
    const __bf16* __restrict__ A,   // [M x K] row-major
    const __bf16* __restrict__ Bm,  // [K x N] row-major
    int K, int N, int bx, int by,
    __bf16* lsA, __bf16* lsB, v8f acc[2][4]) {
  const int tid  = threadIdx.x;
  const int lane = tid & 31;
  const int wv   = tid >> 5;        // 0..7
  const int wm   = wv & 3;          // 4 waves along M (32 rows each)
  const int wn   = wv >> 2;         // 2 waves along N (64 cols each)
  const int hi   = (lane >> 4) & 1;
  const int ln16 = lane & 15;

  #pragma unroll
  for (int mt = 0; mt < 2; ++mt)
    #pragma unroll
    for (int nt = 0; nt < 4; ++nt) { v8f z = {}; acc[mt][nt] = z; }

  const int gm0 = bx * 128;
  const int gn0 = by * 128;

  for (int k0 = 0; k0 < K; k0 += 64) {
    __syncthreads();
    // ---- stage A tile 128x64 via async global->LDS B128 (ASYNCcnt) ----
    // 128*64 bf16 = 16 KB = 1024 x 16B chunks; 4 chunks per thread.
    #pragma unroll
    for (int i = 0; i < 4; ++i) {
      int idx = tid + i * 256;            // 0..1023
      int rm  = idx >> 3;                 // row 0..127
      int kp  = (idx & 7) * 8;            // bf16 index, 16B chunks
      async_load_b128(&lsA[rm * ASTR + kp],
                      &A[(size_t)(gm0 + rm) * K + k0 + kp]);
    }
    // ---- stage B tile transposed Bt[n][k]: dword loads, b16 stores ----
    // 64k x 128n = 8192 elems = 4096 dwords; 16 dwords per thread.
    #pragma unroll
    for (int i = 0; i < 16; ++i) {
      int e  = tid + i * 256;             // 0..4095 dword index
      int kk = e >> 6;                    // 0..63
      int nn = (e & 63) * 2;              // n pair
      unsigned int d =
          *(const unsigned int*)(&Bm[(size_t)(k0 + kk) * N + gn0 + nn]);
      lsB[nn * ASTR + kk]       = ((const __bf16*)&d)[0];
      lsB[(nn + 1) * ASTR + kk] = ((const __bf16*)&d)[1];
    }
    wait_asynccnt0();
    __syncthreads();

    #pragma unroll
    for (int ks = 0; ks < 64; ks += 32) {
      FragAB af[2], bfg[4];
      #pragma unroll
      for (int mt = 0; mt < 2; ++mt) {
        int row = wm * 32 + mt * 16 + ln16;
        #pragma unroll
        for (int j = 0; j < 8; ++j)
          af[mt].u[j] =
              *(const unsigned int*)(&lsA[row * ASTR + ks + kbaseA(j, hi)]);
      }
      #pragma unroll
      for (int nt = 0; nt < 4; ++nt) {
        int col = wn * 64 + nt * 16 + ln16;
        #pragma unroll
        for (int j = 0; j < 8; ++j)
          bfg[nt].u[j] =
              *(const unsigned int*)(&lsB[col * ASTR + ks + kbaseB(j, hi)]);
      }
      #pragma unroll
      for (int mt = 0; mt < 2; ++mt)
        #pragma unroll
        for (int nt = 0; nt < 4; ++nt)
          acc[mt][nt] = __builtin_amdgcn_wmma_f32_16x16x32_bf16(
              false, af[mt].v, false, bfg[nt].v, (short)0, acc[mt][nt],
              false, false);
    }
  }
}

// ---------------- QKVG projection: silu(X @ W) scattered per head ----------------
__global__ __launch_bounds__(256) void qkvg_gemm_kernel(
    const __bf16* __restrict__ X, const __bf16* __restrict__ W,
    __bf16* __restrict__ qb, __bf16* __restrict__ kb,
    __bf16* __restrict__ vt, __bf16* __restrict__ gb) {
  __shared__ __bf16 lsA[128 * ASTR];
  __shared__ __bf16 lsB[128 * ASTR];
  v8f acc[2][4];
  gemm_core_128x128(X, W, D_MODEL, 4 * D_MODEL, blockIdx.x, blockIdx.y,
                    lsA, lsB, acc);

  const int tid  = threadIdx.x;
  const int lane = tid & 31;
  const int wv   = tid >> 5, wm = wv & 3, wn = wv >> 2;
  const int hi   = (lane >> 4) & 1, ln16 = lane & 15;
  #pragma unroll
  for (int mt = 0; mt < 2; ++mt)
    #pragma unroll
    for (int nt = 0; nt < 4; ++nt)
      #pragma unroll
      for (int r = 0; r < 8; ++r) {
        int grow = blockIdx.x * 128 + wm * 32 + mt * 16 + r + 8 * hi;
        int gcol = blockIdx.y * 128 + wn * 64 + nt * 16 + ln16;
        float v  = silu_f(acc[mt][nt][r]);
        int which = gcol >> 10;          // 0=q 1=k 2=v 3=g
        int d  = gcol & 1023;
        int h  = d >> 6, dd = d & 63;
        int b  = grow >> 11, s = grow & 2047;
        int bh = b * NUM_HEADS + h;
        __bf16 bv = (__bf16)v;
        if (which == 0)      qb[((size_t)bh * SEQ + s) * HEAD_DIM + dd] = bv;
        else if (which == 1) kb[((size_t)bh * SEQ + s) * HEAD_DIM + dd] = bv;
        else if (which == 2) vt[((size_t)bh * HEAD_DIM + dd) * SEQ + s] = bv; // transposed
        else                 gb[((size_t)bh * SEQ + s) * HEAD_DIM + dd] = bv;
      }
}

// ---------------- streaming SiLU attention + LayerNorm*gate ----------------
#define SST 34   // score tile LDS stride (bf16), even
#define OST 66   // out tile LDS stride (f32)

__global__ __launch_bounds__(128) void silu_attention_kernel(
    const __bf16* __restrict__ qb, const __bf16* __restrict__ kb,
    const __bf16* __restrict__ vt, const __bf16* __restrict__ gb,
    const int* __restrict__ mask, const float* __restrict__ norm,
    const float* __restrict__ gamma, const float* __restrict__ beta,
    __bf16* __restrict__ attn) {
  __shared__ __bf16 sS[4][16 * SST];   // wave-private score repack buffers
  __shared__ float  sO[4][16 * OST];   // wave-private output rows

  const int tid  = threadIdx.x;
  const int lane = tid & 31;
  const int wv   = tid >> 5;                 // 0..3
  const int hi   = (lane >> 4) & 1, ln16 = lane & 15;

  const int bh = blockIdx.x;                 // 0..31
  const int b  = bh >> 4, h = bh & 15;
  const int q0 = blockIdx.y * 64 + wv * 16;  // this wave's q-row base

  const __bf16* qbase = qb + (size_t)bh * SEQ * HEAD_DIM;
  const __bf16* kbp   = kb + (size_t)bh * SEQ * HEAD_DIM;
  const __bf16* vbase = vt + (size_t)bh * HEAD_DIM * SEQ;

  // Q fragment (16 rows x 64 d) = two 16x32 A-frags, held for the whole stream
  FragAB aq[2];
  {
    const __bf16* rp = qbase + (size_t)(q0 + ln16) * HEAD_DIM;
    #pragma unroll
    for (int kk = 0; kk < 2; ++kk)
      #pragma unroll
      for (int j = 0; j < 8; ++j)
        aq[kk].u[j] = *(const unsigned int*)(rp + kk * 32 + kbaseA(j, hi));
  }

  // per-lane normalizers for the 8 C-layout rows
  float nrm[8];
  #pragma unroll
  for (int r = 0; r < 8; ++r) nrm[r] = norm[b * SEQ + q0 + r + 8 * hi];

  v8f oacc[4];
  #pragma unroll
  for (int nt = 0; nt < 4; ++nt) { v8f z = {}; oacc[nt] = z; }

  __bf16* sw = sS[wv];
  const int* mbase = mask + (size_t)b * SEQ * SEQ;

  for (int kt0 = 0; kt0 < SEQ; kt0 += 32) {
    // ---- scores: Q(16x64) . K^T(64x32) via 4 WMMAs ----
    v8f sc[2];
    #pragma unroll
    for (int nn = 0; nn < 2; ++nn) {
      FragAB bk0, bk1;
      const __bf16* kp = kbp + (size_t)(kt0 + nn * 16 + ln16) * HEAD_DIM;
      #pragma unroll
      for (int j = 0; j < 8; ++j) {
        int kbo = kbaseB(j, hi);
        bk0.u[j] = *(const unsigned int*)(kp + kbo);
        bk1.u[j] = *(const unsigned int*)(kp + 32 + kbo);
      }
      v8f z = {};
      sc[nn] = __builtin_amdgcn_wmma_f32_16x16x32_bf16(
          false, aq[0].v, false, bk0.v, (short)0, z, false, false);
      sc[nn] = __builtin_amdgcn_wmma_f32_16x16x32_bf16(
          false, aq[1].v, false, bk1.v, (short)0, sc[nn], false, false);
    }
    // ---- silu * mask / normalizer, repack C-layout -> A-layout via LDS ----
    #pragma unroll
    for (int nn = 0; nn < 2; ++nn) {
      int keyl = nn * 16 + ln16;
      #pragma unroll
      for (int r = 0; r < 8; ++r) {
        int rowl = r + 8 * hi;
        int qg   = q0 + rowl;
        float x  = sc[nn][r] * ATTN_SCALE;
        int   m  = mbase[(size_t)qg * SEQ + kt0 + keyl];
        float v  = m ? (silu_f(x) / nrm[r]) : 0.0f;
        sw[rowl * SST + keyl] = (__bf16)v;   // wave-private, DS ops in-order
      }
    }
    FragAB as_;
    #pragma unroll
    for (int j = 0; j < 8; ++j)
      as_.u[j] = *(const unsigned int*)(&sw[ln16 * SST + kbaseA(j, hi)]);

    // ---- accumulate O += S(16x32) . V(32x64) via 4 WMMAs ----
    #pragma unroll
    for (int nt = 0; nt < 4; ++nt) {
      FragAB bv;
      const __bf16* vp = vbase + (size_t)(nt * 16 + ln16) * SEQ + kt0;
      #pragma unroll
      for (int j = 0; j < 8; ++j)
        bv.u[j] = *(const unsigned int*)(vp + kbaseB(j, hi));
      oacc[nt] = __builtin_amdgcn_wmma_f32_16x16x32_bf16(
          false, as_.v, false, bv.v, (short)0, oacc[nt], false, false);
    }
  }

  // ---- epilogue: LayerNorm over d (=64) then * gate ----
  float* ow = sO[wv];
  #pragma unroll
  for (int nt = 0; nt < 4; ++nt)
    #pragma unroll
    for (int r = 0; r < 8; ++r)
      ow[(r + 8 * hi) * OST + nt * 16 + ln16] = oacc[nt][r];

  if (lane < 16) {
    int rowl = lane;
    int qg   = q0 + rowl;
    float* rp = ow + rowl * OST;
    float mu = 0.f;
    for (int d = 0; d < HEAD_DIM; ++d) mu += rp[d];
    mu *= (1.0f / HEAD_DIM);
    float var = 0.f;
    for (int d = 0; d < HEAD_DIM; ++d) { float t = rp[d] - mu; var += t * t; }
    var *= (1.0f / HEAD_DIM);
    float rstd = rsqrtf(var + LN_EPS);
    const __bf16* gp = gb + ((size_t)bh * SEQ + qg) * HEAD_DIM;
    __bf16* op = attn + (((size_t)b * SEQ + qg) * NUM_HEADS + h) * HEAD_DIM;
    for (int d = 0; d < HEAD_DIM; ++d) {
      float y = (rp[d] - mu) * rstd * gamma[d] + beta[d];
      op[d] = (__bf16)(y * (float)gp[d]);
    }
  }
}

// ---------------- output projection: attn @ W_out -> fp32 ----------------
__global__ __launch_bounds__(256) void outproj_gemm_kernel(
    const __bf16* __restrict__ X, const __bf16* __restrict__ W,
    float* __restrict__ out) {
  __shared__ __bf16 lsA[128 * ASTR];
  __shared__ __bf16 lsB[128 * ASTR];
  v8f acc[2][4];
  gemm_core_128x128(X, W, D_MODEL, D_MODEL, blockIdx.x, blockIdx.y,
                    lsA, lsB, acc);

  const int tid  = threadIdx.x;
  const int lane = tid & 31;
  const int wv   = tid >> 5, wm = wv & 3, wn = wv >> 2;
  const int hi   = (lane >> 4) & 1, ln16 = lane & 15;
  #pragma unroll
  for (int mt = 0; mt < 2; ++mt)
    #pragma unroll
    for (int nt = 0; nt < 4; ++nt)
      #pragma unroll
      for (int r = 0; r < 8; ++r) {
        int grow = blockIdx.x * 128 + wm * 32 + mt * 16 + r + 8 * hi;
        int gcol = blockIdx.y * 128 + wn * 64 + nt * 16 + ln16;
        out[(size_t)grow * D_MODEL + gcol] = acc[mt][nt][r];
      }
}

// ---------------- host-side launch ----------------
extern "C" void kernel_launch(void* const* d_in, const int* in_sizes, int n_in,
                              void* d_out, int out_size, void* d_ws, size_t ws_size,
                              hipStream_t stream) {
  const float* tokens = (const float*)d_in[0];
  const int*   mask   = (const int*)d_in[1];
  const float* Wqkuv  = (const float*)d_in[2];
  const float* Wout   = (const float*)d_in[3];
  const float* gamma  = (const float*)d_in[4];
  const float* beta   = (const float*)d_in[5];
  float* out = (float*)d_out;

  // workspace layout (bytes), all 256-aligned; total ~58 MB
  char* ws = (char*)d_ws;
  const size_t SZ_TOK  = (size_t)ROWS * D_MODEL * 2;        // 8 MB
  const size_t SZ_WQ   = (size_t)D_MODEL * 4 * D_MODEL * 2; // 8 MB
  const size_t SZ_WO   = (size_t)D_MODEL * D_MODEL * 2;     // 2 MB
  const size_t SZ_HEAD = (size_t)BATCH * NUM_HEADS * SEQ * HEAD_DIM * 2; // 8 MB
  size_t off = 0;
  __bf16* tok_bf   = (__bf16*)(ws + off); off += SZ_TOK;
  __bf16* wqkuv_bf = (__bf16*)(ws + off); off += SZ_WQ;
  __bf16* wout_bf  = (__bf16*)(ws + off); off += SZ_WO;
  __bf16* qb       = (__bf16*)(ws + off); off += SZ_HEAD;
  __bf16* kb       = (__bf16*)(ws + off); off += SZ_HEAD;
  __bf16* vt       = (__bf16*)(ws + off); off += SZ_HEAD;   // [B,H,64,S]
  __bf16* gb       = (__bf16*)(ws + off); off += SZ_HEAD;
  __bf16* attn_bf  = (__bf16*)(ws + off); off += SZ_TOK;    // [B,S,D]
  float*  normf    = (float*)(ws + off);  off += (size_t)ROWS * 4;
  (void)ws_size; (void)in_sizes; (void)n_in; (void)out_size;

  // 1) converts
  {
    int n = ROWS * D_MODEL;                     // 4,194,304
    cvt_f32_bf16_kernel<<<(n + 255) / 256, 256, 0, stream>>>(tokens, tok_bf, n);
    int nw = D_MODEL * 4 * D_MODEL;             // 4,194,304
    cvt_f32_bf16_kernel<<<(nw + 255) / 256, 256, 0, stream>>>(Wqkuv, wqkuv_bf, nw);
    int no = D_MODEL * D_MODEL;                 // 1,048,576
    cvt_f32_bf16_kernel<<<(no + 255) / 256, 256, 0, stream>>>(Wout, wout_bf, no);
  }
  // 2) mask normalizer
  mask_norm_kernel<<<(ROWS + 255) / 256, 256, 0, stream>>>(mask, normf);
  // 3) QKVG projection (4096 x 4096 x 1024)
  qkvg_gemm_kernel<<<dim3(ROWS / 128, (4 * D_MODEL) / 128), 256, 0, stream>>>(
      tok_bf, wqkuv_bf, qb, kb, vt, gb);
  // 4) streaming SiLU attention + LN*gate
  silu_attention_kernel<<<dim3(BATCH * NUM_HEADS, SEQ / 64), 128, 0, stream>>>(
      qb, kb, vt, gb, mask, normf, gamma, beta, attn_bf);
  // 5) output projection (4096 x 1024 x 1024) -> fp32
  outproj_gemm_kernel<<<dim3(ROWS / 128, D_MODEL / 128), 256, 0, stream>>>(
      attn_bf, wout_bf, out);
}
